// EDUEncoder_30640296690272
// MI455X (gfx1250) — matrix operand order repaired and checked by reading, not compiled, not run on hardware
//
#include <hip/hip_runtime.h>
#include <hip/hip_bf16.h>

#define H   512
#define T   512
#define BB  32
#define GH  1536   // 3*H

#define USE_TDM 1
#define LDS_STRIDE 520          // 512 + 8 bf16 pad -> 1040B rows (bank-skew 4)
#define SCAN_BLOCKS 32

typedef __attribute__((ext_vector_type(16))) __bf16 v16bf;
typedef __attribute__((ext_vector_type(8)))  float  v8f;
typedef __attribute__((ext_vector_type(4)))  unsigned int v4u;
typedef __attribute__((ext_vector_type(4)))  unsigned int u32x4;
typedef __attribute__((ext_vector_type(8)))  int i32x8;
typedef __attribute__((ext_vector_type(4)))  int i32x4;

union ABf { v4u u[2]; v16bf v; };

// ---------------- fragment loaders (contiguous k = 16*(L/16)+s, same order for A and B) --------

// A fragment (16x32 bf16, row-major): lane L -> row = L%16,
// K = (L/16)*16..+15 contiguous (two 16B chunks).
__device__ inline v16bf load_frag_a(const __bf16* base, int rowstride, int lane) {
    int r  = lane & 15;
    int kb = (lane >> 4) * 16;
    const __bf16* p = base + (size_t)r * rowstride + kb;
    ABf f;
    f.u[0] = *(const v4u*)(p);
    f.u[1] = *(const v4u*)(p + 8);
    return f.v;
}

// B fragment (32x16 bf16): lane L -> column N = L%16, K = (L/16)*16..+15
// contiguous (PyTorch W[g,k] row-major).
__device__ inline v16bf load_frag_b(const __bf16* base, int rowstride, int lane) {
    int g  = lane & 15;
    int kh = (lane >> 4) * 16;
    const __bf16* p = base + (size_t)g * rowstride + kh;
    ABf f;
    f.u[0] = *(const v4u*)(p);
    f.u[1] = *(const v4u*)(p + 8);
    return f.v;
}

__device__ inline v8f wmma_bf16(v16bf a, v16bf b, v8f c) {
    return __builtin_amdgcn_wmma_f32_16x16x32_bf16(
        false, a, false, b, (short)0, c, false, false);
}

__device__ inline float sigmoidf_(float x) { return 1.0f / (1.0f + __expf(-x)); }
__device__ inline float tanhf_(float x) {
    float e = __expf(-2.0f * x);
    return (1.0f - e) / (1.0f + e);
}

// ---------------- Tensor Data Mover: 16x512 bf16 row-slab -> LDS ------------
// D# per CDNA5 ISA: 2D tile, data_size=2B,
// pad_enable: after every 256 dwords (one 512-elem row) skip 4 dwords (16B)
// -> LDS row stride 1040B = LDS_STRIDE elements.
#if USE_TDM
__device__ inline void tdm_load_slab(const __bf16* gsrc, unsigned lds_off) {
    unsigned long long ga = (unsigned long long)(size_t)gsrc;
    u32x4 g0;
    g0[0] = 1u;                                  // count=1 (valid), user mode
    g0[1] = lds_off;                             // lds_addr (bytes)
    g0[2] = (unsigned)(ga & 0xFFFFFFFFu);        // global_addr lo
    g0[3] = (unsigned)((ga >> 32) & 0x01FFFFFFu) // global_addr hi (57-bit)
            | 0x80000000u;                       // type=2 ("image") bits 127:126
    i32x8 g1;
    g1[0] = (1 << 16)      // data_size = 2B
          | (1 << 20)      // pad_enable
          | (7 << 22)      // pad_interval code 7 = 256 dwords
          | (3 << 25);     // pad_amount  code 3 = 4 dwords (16B)
    g1[1] = (512 << 16);   // tensor_dim0 lo16 = 512
    g1[2] = (16 << 16);    // tensor_dim0 hi = 0 | tensor_dim1 lo16 = 16
    g1[3] = (512 << 16);   // tensor_dim1 hi = 0 | tile_dim0 = 512
    g1[4] = 16;            // tile_dim1 = 16 | tile_dim2 = 0
    g1[5] = 512;           // tensor_dim0_stride lo32 = 512 (elements)
    g1[6] = 0;
    g1[7] = 0;
    i32x4 z4 = {0, 0, 0, 0};
#if __clang_major__ >= 23
    i32x8 z8 = {0, 0, 0, 0, 0, 0, 0, 0};
    __builtin_amdgcn_tensor_load_to_lds(g0, g1, z4, z4, z8, 0);
#else
    __builtin_amdgcn_tensor_load_to_lds(g0, g1, z4, z4, 0);
#endif
}
#endif

// ---------------- prep kernels ----------------------------------------------

__global__ void cast_f32_bf16(const float* __restrict__ src,
                              __bf16* __restrict__ dst, int n) {
    int i = blockIdx.x * blockDim.x + threadIdx.x;
    if (i < n) dst[i] = (__bf16)src[i];
}

// x: [B, T, H] f32  ->  dst: [T*B, H] bf16 with row = t*B + b (time-major)
__global__ void transpose_cast_x(const float* __restrict__ x,
                                 __bf16* __restrict__ dst) {
    int i = blockIdx.x * blockDim.x + threadIdx.x;
    if (i >= BB * T * H) return;
    int h = i & (H - 1);
    int t = (i >> 9) & (T - 1);
    int b = i >> 18;
    dst[((size_t)t * BB + b) * H + h] = (__bf16)x[i];
}

// ---------------- input-projection GEMM -------------------------------------
// C[M,1536] = A[M,512] * W[1536,512]^T + bias (bf16 in, f32 acc, bf16 out)
// Block 256 (8 waves); WG tile 64(M) x 128(N); wave tile 32x32.
__global__ void gemm_xproj(const __bf16* __restrict__ A,
                           const __bf16* __restrict__ W,
                           const float*  __restrict__ bias,
                           __bf16* __restrict__ C) {
    const int lane = threadIdx.x & 31;
    const int w    = threadIdx.x >> 5;
    const int wm   = w & 1;
    const int wn   = w >> 1;
    const int Mb   = blockIdx.x * 64 + wm * 32;
    const int Nb   = blockIdx.y * 128 + wn * 32;

    v8f c00 = {}, c01 = {}, c10 = {}, c11 = {};
    #pragma unroll 4
    for (int kb = 0; kb < H; kb += 32) {
        v16bf a0 = load_frag_a(A + (size_t)Mb * H + kb, H, lane);
        v16bf a1 = load_frag_a(A + (size_t)(Mb + 16) * H + kb, H, lane);
        v16bf b0 = load_frag_b(W + (size_t)Nb * H + kb, H, lane);
        v16bf b1 = load_frag_b(W + (size_t)(Nb + 16) * H + kb, H, lane);
        c00 = wmma_bf16(a0, b0, c00);
        c01 = wmma_bf16(a0, b1, c01);
        c10 = wmma_bf16(a1, b0, c10);
        c11 = wmma_bf16(a1, b1, c11);
    }

    const int cl = lane & 15;
    const int hi = lane >> 4;
    #pragma unroll
    for (int q = 0; q < 4; ++q) {
        int mb  = Mb + ((q >> 1) ? 16 : 0);
        int col = Nb + ((q & 1) ? 16 : 0) + cl;
        v8f acc = (q == 0) ? c00 : (q == 1) ? c01 : (q == 2) ? c10 : c11;
        float bv = bias[col];
        int rbase = mb + (hi ? 8 : 0);
        #pragma unroll
        for (int i = 0; i < 8; ++i)
            C[(size_t)(rbase + i) * GH + col] = (__bf16)(acc[i] + bv);
    }
}

// ---------------- recurrent scan --------------------------------------------
// Persistent: 32 blocks x 64 threads (2 waves).  Both waves of a block share
// one 16-column slice c0 = blockIdx.x*16 (wave = batch half).  The block's 48
// W_hh gate-rows live in LDS (TDM-preloaded, bank-skewed).  Per step: WMMA
// r/z/n accum chains (K=512), wave-local gate math, masked h/y writes, then a
// grid-wide atomic barrier over double-buffered hidden state.
__global__ __launch_bounds__(64)
void gru_scan(const __bf16* __restrict__ xproj,  // [T*B, 3H] bf16
              const __bf16* __restrict__ Whh,    // [3H, H]  bf16
              const float*  __restrict__ bhh,    // [3H]
              const int*    __restrict__ len,    // [B]
              float*  __restrict__ h32,          // [2][B][H]
              __bf16* __restrict__ h16,          // [2][B][H]
              __bf16* __restrict__ y16,          // [T*B, H] (layer 0)
              float*  __restrict__ yout,         // [B, T, H] (layer 1)
              unsigned int* __restrict__ cnt,    // [T]
              int layer) {
    __shared__ __bf16 ldsW[3 * 16 * LDS_STRIDE];   // ~49.9 KB

    const int lane = threadIdx.x & 31;
    const int m    = threadIdx.x >> 5;            // batch half (wave id)
    const int c0   = blockIdx.x * 16;             // hidden-column base
    const int col  = c0 + (lane & 15);
    const int rb   = m * 16 + ((lane >> 4) ? 8 : 0);

    // ---- preload this block's W_hh rows into LDS ----
#if USE_TDM
    if (threadIdx.x < 32) {                       // wave 0 issues TDM
        #pragma unroll
        for (int g = 0; g < 3; ++g)
            tdm_load_slab(Whh + (size_t)(g * H + c0) * H,
                          (unsigned)(size_t)(&ldsW[g * 16 * LDS_STRIDE]));
        __builtin_amdgcn_s_wait_tensorcnt(0);
    }
#else
    for (int idx = threadIdx.x; idx < 3 * 16 * (H / 8); idx += 64) {
        int row = idx >> 6;                       // 0..47
        int ch  = idx & 63;                       // 16B chunk in row
        int gate = row >> 4, rr = row & 15;
        v4u d = *(const v4u*)(Whh + (size_t)(gate * H + c0 + rr) * H + ch * 8);
        *(v4u*)(&ldsW[row * LDS_STRIDE + ch * 8]) = d;
    }
#endif
    __syncthreads();

    const float br = bhh[col];
    const float bz = bhh[H + col];
    const float bn = bhh[2 * H + col];
    int lens[8];
    #pragma unroll
    for (int i = 0; i < 8; ++i) lens[i] = len[rb + i];

    for (int t = 0; t < T; ++t) {
        const int p = t & 1, q = p ^ 1;
        const __bf16* hA = h16 + (size_t)p * BB * H;

        // prefetch next step's x_proj gate rows
        if (t + 1 < T) {
            const __bf16* xn_ = xproj + (size_t)((t + 1) * BB) * GH
                              + (size_t)rb * GH + col;
            __builtin_prefetch(xn_, 0, 1);
            __builtin_prefetch(xn_ + H, 0, 1);
            __builtin_prefetch(xn_ + 2 * H, 0, 1);
        }

        v8f cr = {}, cz = {}, cn = {};
        #pragma unroll 4
        for (int kb = 0; kb < H; kb += 32) {
            v16bf a   = load_frag_a(hA + (size_t)(m * 16) * H + kb, H, lane);
            v16bf brf = load_frag_b(&ldsW[0 * 16 * LDS_STRIDE + kb], LDS_STRIDE, lane);
            v16bf bzf = load_frag_b(&ldsW[1 * 16 * LDS_STRIDE + kb], LDS_STRIDE, lane);
            v16bf bnf = load_frag_b(&ldsW[2 * 16 * LDS_STRIDE + kb], LDS_STRIDE, lane);
            cr = wmma_bf16(a, brf, cr);
            cz = wmma_bf16(a, bzf, cz);
            cn = wmma_bf16(a, bnf, cn);
        }

        const __bf16* xrow = xproj + (size_t)(t * BB) * GH;
        const float*  h32p = h32 + (size_t)p * BB * H;
        float*        h32q = h32 + (size_t)q * BB * H;
        __bf16*       h16q = h16 + (size_t)q * BB * H;

        #pragma unroll
        for (int i = 0; i < 8; ++i) {
            int b = rb + i;
            float xr = (float)xrow[(size_t)b * GH + col];
            float xz = (float)xrow[(size_t)b * GH + H + col];
            float xn = (float)xrow[(size_t)b * GH + 2 * H + col];
            float r = sigmoidf_(xr + cr[i] + br);
            float z = sigmoidf_(xz + cz[i] + bz);
            float n = tanhf_(xn + r * (cn[i] + bn));
            float hp   = h32p[(size_t)b * H + col];
            float hnew = (1.0f - z) * n + z * hp;
            bool  mk   = t < lens[i];
            float hkeep = mk ? hnew : hp;
            h32q[(size_t)b * H + col] = hkeep;
            h16q[(size_t)b * H + col] = (__bf16)hkeep;
            float yv = mk ? hnew : 0.0f;
            if (layer == 0)
                y16[((size_t)(t * BB) + b) * H + col] = (__bf16)yv;
            else
                yout[(size_t)b * T * H + (size_t)t * H + col] = yv;
        }

        // --- grid-wide barrier (release h writes / acquire next reads) ---
        __threadfence();
        __syncthreads();
        if (threadIdx.x == 0) {
            __hip_atomic_fetch_add(&cnt[t], 1u, __ATOMIC_ACQ_REL,
                                   __HIP_MEMORY_SCOPE_AGENT);
            while (__hip_atomic_load(&cnt[t], __ATOMIC_ACQUIRE,
                                     __HIP_MEMORY_SCOPE_AGENT) < gridDim.x) {}
        }
        __syncthreads();
        __threadfence();
    }
}

// ---------------- host launcher ---------------------------------------------

extern "C" void kernel_launch(void* const* d_in, const int* in_sizes, int n_in,
                              void* d_out, int out_size, void* d_ws, size_t ws_size,
                              hipStream_t stream) {
    const float* edu_hidden = (const float*)d_in[0];
    const int*   lengths    = (const int*)d_in[1];
    const float* W_ih_0 = (const float*)d_in[2];
    const float* W_hh_0 = (const float*)d_in[3];
    const float* b_ih_0 = (const float*)d_in[4];
    const float* b_hh_0 = (const float*)d_in[5];
    const float* W_ih_1 = (const float*)d_in[6];
    const float* W_hh_1 = (const float*)d_in[7];
    const float* b_ih_1 = (const float*)d_in[8];
    const float* b_hh_1 = (const float*)d_in[9];
    float* out = (float*)d_out;

    char* base = (char*)d_ws;
    size_t cur = 0;
    auto alloc = [&](size_t bytes) -> void* {
        void* p = base + cur;
        cur += (bytes + 255) & ~(size_t)255;
        return p;
    };

    unsigned int* cnt0 = (unsigned int*)alloc(T * 4);
    unsigned int* cnt1 = (unsigned int*)alloc(T * 4);
    float*  h32   = (float*)alloc(2 * BB * H * 4);
    __bf16* h16   = (__bf16*)alloc(2 * BB * H * 2);
    size_t zero_region_end = cur;             // counters + h buffers

    __bf16* xA16   = (__bf16*)alloc((size_t)T * BB * H * 2);
    __bf16* y16    = (__bf16*)alloc((size_t)T * BB * H * 2);
    __bf16* wih0   = (__bf16*)alloc((size_t)GH * H * 2);
    __bf16* whh0   = (__bf16*)alloc((size_t)GH * H * 2);
    __bf16* wih1   = (__bf16*)alloc((size_t)GH * H * 2);
    __bf16* whh1   = (__bf16*)alloc((size_t)GH * H * 2);
    __bf16* xproj  = (__bf16*)alloc((size_t)T * BB * GH * 2);
    (void)ws_size; (void)n_in; (void)in_sizes; (void)out_size;

    // zero barrier counters + hidden-state double buffers
    hipMemsetAsync(d_ws, 0, zero_region_end, stream);

    // bf16 conversions
    {
        int n = BB * T * H;
        transpose_cast_x<<<(n + 255) / 256, 256, 0, stream>>>(edu_hidden, xA16);
        int nw = GH * H;
        cast_f32_bf16<<<(nw + 255) / 256, 256, 0, stream>>>(W_ih_0, wih0, nw);
        cast_f32_bf16<<<(nw + 255) / 256, 256, 0, stream>>>(W_hh_0, whh0, nw);
        cast_f32_bf16<<<(nw + 255) / 256, 256, 0, stream>>>(W_ih_1, wih1, nw);
        cast_f32_bf16<<<(nw + 255) / 256, 256, 0, stream>>>(W_hh_1, whh1, nw);
    }

    dim3 ggrid(T * BB / 64, GH / 128);

    // ---- layer 0 ----
    gemm_xproj<<<ggrid, 256, 0, stream>>>(xA16, wih0, b_ih_0, xproj);
    gru_scan<<<SCAN_BLOCKS, 64, 0, stream>>>(xproj, whh0, b_hh_0, lengths,
                                             h32, h16, y16, nullptr, cnt0, 0);

    // re-zero hidden state for layer 1
    hipMemsetAsync(h32, 0, (size_t)2 * BB * H * 4, stream);
    hipMemsetAsync(h16, 0, (size_t)2 * BB * H * 2, stream);

    // ---- layer 1 ----
    gemm_xproj<<<ggrid, 256, 0, stream>>>(y16, wih1, b_ih_1, xproj);
    gru_scan<<<SCAN_BLOCKS, 64, 0, stream>>>(xproj, whh1, b_hh_1, lengths,
                                             h32, h16, nullptr, out, cnt1, 1);
}
